// Attention_4105988735559
// MI455X (gfx1250) — compile-verified, matrix operand order checked by reading, not run on hardware
//
#include <hip/hip_runtime.h>

// ---------------- problem constants ----------------
constexpr int B   = 2;
constexpr int SQ  = 2048;
constexpr int SKV = 2048;
constexpr int H   = 1024;
constexpr int NH  = 16;
constexpr int HD  = 64;

typedef __attribute__((ext_vector_type(16))) _Float16 v16h;
typedef __attribute__((ext_vector_type(8)))  _Float16 v8h;
typedef __attribute__((ext_vector_type(8)))  float    v8f;
typedef __attribute__((ext_vector_type(4)))  unsigned int v4u;
typedef __attribute__((ext_vector_type(8)))  int v8i;
typedef __attribute__((ext_vector_type(4)))  int v4i;

// D = A(16x32) * B(32x16) + C, f16 in / f32 acc (probe-verified builtin)
__device__ __forceinline__ v8f wmma16(v16h a, v16h b, v8f c) {
  return __builtin_amdgcn_wmma_f32_16x16x32_f16(
      false, a, false, b, (short)0, c, false, false);
}

// 16-bit A/B fragment per CDNA5 ISA: elems 0..7 = K base+0..7,
// elems 8..15 = K 16+base+0..7 (base = 8 for lanes 16..31).
__device__ __forceinline__ v16h frag_from_f16(const _Float16* __restrict__ p0,
                                              const _Float16* __restrict__ p1) {
  v8h a = *(const v8h*)p0;
  v8h b = *(const v8h*)p1;
  v16h v;
#pragma unroll
  for (int i = 0; i < 8; ++i) { v[i] = a[i]; v[i + 8] = b[i]; }
  return v;
}

// =====================================================================
// Kernel 0: bulk f32 -> f16 conversion (keeps cvt out of GEMM hot loops)
// =====================================================================
__global__ __launch_bounds__(256) void cvt_kernel(const float* __restrict__ src,
                                                  _Float16* __restrict__ dst,
                                                  int n) {
  const int i = (blockIdx.x * 256 + threadIdx.x) * 8;
  if (i + 8 <= n) {
#pragma unroll
    for (int e = 0; e < 8; ++e) dst[i + e] = (_Float16)src[i + e];
  }
}

// =====================================================================
// Kernel 1: QKV projections, all-f16 operands.
// Wave tile 32x64 (8 WMMAs/k-step); block = 8 waves = 64x256 tile.
// Grid: (4096/64, 1024/256, 3)   z: 0=Q(bias,scale) 1=K 2=V(bias,transposed)
// =====================================================================
__global__ __launch_bounds__(256) void qkv_proj_kernel(
    const _Float16* __restrict__ Xq, const _Float16* __restrict__ Xkv,
    const _Float16* __restrict__ Wqh, const _Float16* __restrict__ Wkh,
    const _Float16* __restrict__ Wvh,
    const float* __restrict__ bq, const float* __restrict__ bv,
    _Float16* __restrict__ Qh,   // [B,NH,SQ,HD], pre-scaled by 1/8
    _Float16* __restrict__ Kh,   // [B,NH,SKV,HD]
    _Float16* __restrict__ VTh)  // [B,NH,HD,SKV]
{
  const int mode = blockIdx.z;
  const int lane = threadIdx.x & 31;
  const int wave = threadIdx.x >> 5;
  const int hi = lane >> 4, ln = lane & 15, kb = hi * 8;

  const int m0 = blockIdx.x * 64  + (wave & 1) * 32;
  const int n0 = blockIdx.y * 256 + (wave >> 1) * 64;

  const _Float16* X = (mode == 0) ? Xq : Xkv;
  const _Float16* W = (mode == 0) ? Wqh : ((mode == 1) ? Wkh : Wvh);

  const _Float16* ar0 = X + (size_t)(m0 + ln) * H;
  const _Float16* ar1 = X + (size_t)(m0 + 16 + ln) * H;

  v8f acc[2][4];
#pragma unroll
  for (int u = 0; u < 2; ++u)
#pragma unroll
    for (int t = 0; t < 4; ++t) acc[u][t] = v8f{};

  for (int k0 = 0; k0 < H; k0 += 32) {
    v16h a0 = frag_from_f16(ar0 + k0 + kb, ar0 + k0 + 16 + kb);
    v16h a1 = frag_from_f16(ar1 + k0 + kb, ar1 + k0 + 16 + kb);
#pragma unroll
    for (int t = 0; t < 4; ++t) {
      const _Float16* bp = W + (size_t)(n0 + t * 16 + ln) * H + k0 + kb;
      v16h bf = frag_from_f16(bp, bp + 16);
      acc[0][t] = wmma16(a0, bf, acc[0][t]);
      acc[1][t] = wmma16(a1, bf, acc[1][t]);
    }
  }

#pragma unroll
  for (int u = 0; u < 2; ++u) {
#pragma unroll
    for (int t = 0; t < 4; ++t) {
      const int n = n0 + t * 16 + ln;
      const int h = n >> 6, d = n & 63;
#pragma unroll
      for (int r = 0; r < 8; ++r) {
        const int m = m0 + u * 16 + r + hi * 8;
        const int b = m >> 11, s = m & 2047;
        float v = acc[u][t][r];
        if (mode == 0) {
          v = (v + bq[n]) * 0.125f;
          Qh[((size_t)(b * NH + h) * SQ + s) * HD + d] = (_Float16)v;
        } else if (mode == 1) {
          Kh[((size_t)(b * NH + h) * SKV + s) * HD + d] = (_Float16)v;
        } else {
          v += bv[n];
          VTh[((size_t)(b * NH + h) * HD + d) * SKV + s] = (_Float16)v;
        }
      }
    }
  }
}

// ---------------- attention helpers ----------------
__device__ __forceinline__ void softmax_update(v8f& S0, v8f& S1,
                                               float* mrow, float* lrow,
                                               float* alpha,
                                               const float* mtile,
                                               int hi, int ln) {
#pragma unroll
  for (int r = 0; r < 8; ++r) {
    const int row = r + hi * 8;
    S0[r] += (1.0f - mtile[row * 32 + ln])      * -10000.0f;
    S1[r] += (1.0f - mtile[row * 32 + 16 + ln]) * -10000.0f;

    float mx = fmaxf(S0[r], S1[r]);
#pragma unroll
    for (int off = 8; off >= 1; off >>= 1)
      mx = fmaxf(mx, __shfl_xor(mx, off, 16));
    const float mn = fmaxf(mrow[r], mx);
    alpha[r] = __expf(mrow[r] - mn);
    const float p0 = __expf(S0[r] - mn);
    const float p1 = __expf(S1[r] - mn);
    float rs = p0 + p1;
#pragma unroll
    for (int off = 8; off >= 1; off >>= 1)
      rs += __shfl_xor(rs, off, 16);
    lrow[r] = lrow[r] * alpha[r] + rs;
    mrow[r] = mn;
    S0[r] = p0; S1[r] = p1;
  }
}

// C-layout P (f32) -> A-layout f16 fragment via per-wave LDS staging.
__device__ __forceinline__ v16h stage_p(const v8f& S0, const v8f& S1,
                                        _Float16* pb, int hi, int ln, int kb) {
#pragma unroll
  for (int r = 0; r < 8; ++r) {
    const int row = r + hi * 8;
    pb[row * 32 + ln]      = (_Float16)S0[r];
    pb[row * 32 + 16 + ln] = (_Float16)S1[r];
  }
  asm volatile("s_wait_dscnt 0" ::: "memory");  // wave-local LDS RAW
  return frag_from_f16(pb + ln * 32 + kb, pb + ln * 32 + 16 + kb);
}

// =====================================================================
// Kernel 2: flash attention. Wave owns 32 q rows (two 16-row tiles sharing
// K/V fragments -> 16 WMMAs per 32-wide kv step). Block = 8 waves = 8 heads
// of one (b, 32-row q slab); the 32x32 mask tile is staged once per kv step
// with the Tensor Data Mover and reused by all 8 heads.
// Grid: (SQ/32, B*2).
// =====================================================================
__global__ __launch_bounds__(256) void attn_kernel(
    const _Float16* __restrict__ Qh, const _Float16* __restrict__ Kh,
    const _Float16* __restrict__ VTh, const float* __restrict__ mask,
    _Float16* __restrict__ ctx)    // [B,SQ,H] f16
{
  __shared__ float    lds_mask[32 * 32];          // 4 KB mask tile / block
  __shared__ _Float16 lds_p[8][2 * 16 * 32];      // 2 KB P staging / wave

  const int tid  = threadIdx.x;
  const int lane = tid & 31;
  const int wave = tid >> 5;
  const int hi = lane >> 4, ln = lane & 15, kb = hi * 8;

  const int b  = blockIdx.y >> 1;
  const int h  = ((blockIdx.y & 1) << 3) + wave;
  const int bh = b * NH + h;
  const int q0 = blockIdx.x * 32;

  const _Float16* Qp = Qh  + (size_t)bh * SQ * HD;
  const _Float16* Kp = Kh  + (size_t)bh * SKV * HD;
  const _Float16* Vp = VTh + (size_t)bh * HD * SKV;
  const float*    Mp = mask + ((size_t)b * SQ + q0) * SKV;

  // resident Q fragments for both 16-row tiles
  const _Float16* qrA = Qp + (size_t)(q0 + ln) * HD;
  const _Float16* qrB = Qp + (size_t)(q0 + 16 + ln) * HD;
  const v16h QA0 = frag_from_f16(qrA + kb,      qrA + 16 + kb);
  const v16h QA1 = frag_from_f16(qrA + 32 + kb, qrA + 48 + kb);
  const v16h QB0 = frag_from_f16(qrB + kb,      qrB + 16 + kb);
  const v16h QB1 = frag_from_f16(qrB + 32 + kb, qrB + 48 + kb);

  v8f OA[4], OB[4];
  float mA[8], lA[8], mB[8], lB[8];
#pragma unroll
  for (int r = 0; r < 8; ++r) {
    mA[r] = mB[r] = -1e30f; lA[r] = lB[r] = 0.0f;
  }
#pragma unroll
  for (int t = 0; t < 4; ++t) { OA[t] = v8f{}; OB[t] = v8f{}; }

  _Float16* pbA = lds_p[wave];
  _Float16* pbB = lds_p[wave] + 16 * 32;

#if __has_builtin(__builtin_amdgcn_tensor_load_to_lds)
  // Tensor DMA descriptor (CDNA5 ISA §8.3/8.4): 2D tile 32(cols f32) x 32(rows)
  // from the [SQ,SKV] f32 mask plane of batch b, into lds_mask.
  const unsigned long long mbase = (unsigned long long)(uintptr_t)Mp;
  const unsigned ldsaddr = (unsigned)(uintptr_t)(void*)lds_mask;  // LDS offset in low 32b
  v8i g1;
  g1[0] = 0x20000;                 // workgroup_mask=0, data_size=2 (4B)
  g1[1] = (int)((SKV & 0xFFFF) << 16);   // tensor_dim0 low16 (bits 79:48)
  g1[2] = (int)((SQ & 0xFFFF) << 16);    // dim0 hi16 =0 | tensor_dim1 low16
  g1[3] = (int)(32 << 16);               // dim1 hi16 =0 | tile_dim0 = 32
  g1[4] = 32;                            // tile_dim1 = 32, tile_dim2 = 0
  g1[5] = SKV;                           // tensor_dim0_stride low32
  g1[6] = 0;                             // stride hi16 | dim1_stride low16
  g1[7] = 0;
  const v4i gz4 = {0, 0, 0, 0};          // groups 2/3 unused (2D tensor)
  const v8i gz8 = {0, 0, 0, 0, 0, 0, 0, 0};
#endif

  for (int j = 0; j < SKV; j += 32) {
#if __has_builtin(__builtin_amdgcn_tensor_load_to_lds)
    if (wave == 0) {
      const unsigned long long ga = mbase + (unsigned long long)j * 4u;
      v4u g0;
      g0[0] = 1u;                                  // count=1, user descriptor
      g0[1] = ldsaddr;                             // lds_addr
      g0[2] = (unsigned)ga;                        // global_addr[31:0]
      g0[3] = (unsigned)((ga >> 32) & 0x1FFFFFFu)  // global_addr[56:32]
              | (2u << 30);                        // type=2 ("image")
      // 6-arg form (clang-23 / therock-10.0 headers)
      __builtin_amdgcn_tensor_load_to_lds(g0, g1, gz4, gz4, gz8, 0);
      __builtin_amdgcn_s_wait_tensorcnt((short)0);
    }
#else
    for (int i = tid; i < 32 * 32; i += 256)
      lds_mask[i] = Mp[(size_t)(i >> 5) * SKV + j + (i & 31)];
#endif
    __syncthreads();

    // ---- K fragments (shared by both q tiles) + prefetch next kv step ----
    const _Float16* kr0 = Kp + (size_t)(j + ln) * HD;
    const _Float16* kr1 = Kp + (size_t)(j + 16 + ln) * HD;
    if (j + 32 < SKV) {
      __builtin_prefetch(kr0 + 32 * HD, 0, 1);   // global_prefetch_b8
      __builtin_prefetch(kr1 + 32 * HD, 0, 1);
    }
    v16h B0a = frag_from_f16(kr0 + kb,      kr0 + 16 + kb);
    v16h B0b = frag_from_f16(kr0 + 32 + kb, kr0 + 48 + kb);
    v16h B1a = frag_from_f16(kr1 + kb,      kr1 + 16 + kb);
    v16h B1b = frag_from_f16(kr1 + 32 + kb, kr1 + 48 + kb);

    // ---- scores for both q tiles (8 WMMAs) ----
    v8f SA0 = wmma16(QA0, B0a, v8f{}); SA0 = wmma16(QA1, B0b, SA0);
    v8f SA1 = wmma16(QA0, B1a, v8f{}); SA1 = wmma16(QA1, B1b, SA1);
    v8f SB0 = wmma16(QB0, B0a, v8f{}); SB0 = wmma16(QB1, B0b, SB0);
    v8f SB1 = wmma16(QB0, B1a, v8f{}); SB1 = wmma16(QB1, B1b, SB1);

    // ---- mask + online softmax ----
    float aA[8], aB[8];
    softmax_update(SA0, SA1, mA, lA, aA, lds_mask,            hi, ln);
    softmax_update(SB0, SB1, mB, lB, aB, lds_mask + 16 * 32,  hi, ln);

    // ---- P -> A-layout fragments via LDS ----
    v16h PfA = stage_p(SA0, SA1, pbA, hi, ln, kb);
    v16h PfB = stage_p(SB0, SB1, pbB, hi, ln, kb);

    // ---- O = O*alpha + P * V^T (8 WMMAs, V fragments shared) ----
#pragma unroll
    for (int t = 0; t < 4; ++t) {
      const _Float16* vr = Vp + (size_t)(t * 16 + ln) * SKV + j;
      if (j + 32 < SKV) __builtin_prefetch(vr + 32, 0, 1);
      v16h Vf = frag_from_f16(vr + kb, vr + 16 + kb);
#pragma unroll
      for (int r = 0; r < 8; ++r) { OA[t][r] *= aA[r]; OB[t][r] *= aB[r]; }
      OA[t] = wmma16(PfA, Vf, OA[t]);
      OB[t] = wmma16(PfB, Vf, OB[t]);
    }
    __syncthreads();   // protect lds_mask for next kv step
  }

  // ---- normalize and store ctx (f16) ----
  float riA[8], riB[8];
#pragma unroll
  for (int r = 0; r < 8; ++r) { riA[r] = 1.0f / lA[r]; riB[r] = 1.0f / lB[r]; }
#pragma unroll
  for (int t = 0; t < 4; ++t) {
    const int d = t * 16 + ln;
#pragma unroll
    for (int r = 0; r < 8; ++r) {
      const int sA = q0 + r + hi * 8;
      const int sB = sA + 16;
      ctx[((size_t)(b * SQ + sA)) * H + h * HD + d] = (_Float16)(OA[t][r] * riA[r]);
      ctx[((size_t)(b * SQ + sB)) * H + h * HD + d] = (_Float16)(OB[t][r] * riB[r]);
    }
  }
}

// =====================================================================
// Kernel 3: out = ctx @ Wo^T + bo (all-f16 operands, f32 out).
// Same tiling as kernel 1.  Grid: (64, 4).
// =====================================================================
__global__ __launch_bounds__(256) void out_proj_kernel(
    const _Float16* __restrict__ ctxp, const _Float16* __restrict__ Woh,
    const float* __restrict__ bo, float* __restrict__ out)
{
  const int lane = threadIdx.x & 31;
  const int wave = threadIdx.x >> 5;
  const int hi = lane >> 4, ln = lane & 15, kb = hi * 8;

  const int m0 = blockIdx.x * 64  + (wave & 1) * 32;
  const int n0 = blockIdx.y * 256 + (wave >> 1) * 64;

  const _Float16* ar0 = ctxp + (size_t)(m0 + ln) * H;
  const _Float16* ar1 = ctxp + (size_t)(m0 + 16 + ln) * H;

  v8f acc[2][4];
#pragma unroll
  for (int u = 0; u < 2; ++u)
#pragma unroll
    for (int t = 0; t < 4; ++t) acc[u][t] = v8f{};

  for (int k0 = 0; k0 < H; k0 += 32) {
    v16h a0 = frag_from_f16(ar0 + k0 + kb, ar0 + k0 + 16 + kb);
    v16h a1 = frag_from_f16(ar1 + k0 + kb, ar1 + k0 + 16 + kb);
#pragma unroll
    for (int t = 0; t < 4; ++t) {
      const _Float16* bp = Woh + (size_t)(n0 + t * 16 + ln) * H + k0 + kb;
      v16h bf = frag_from_f16(bp, bp + 16);
      acc[0][t] = wmma16(a0, bf, acc[0][t]);
      acc[1][t] = wmma16(a1, bf, acc[1][t]);
    }
  }

#pragma unroll
  for (int u = 0; u < 2; ++u)
#pragma unroll
    for (int t = 0; t < 4; ++t) {
      const int n = n0 + t * 16 + ln;
#pragma unroll
      for (int r = 0; r < 8; ++r) {
        const int m = m0 + u * 16 + r + hi * 8;
        out[(size_t)m * H + n] = acc[u][t][r] + bo[n];
      }
    }
}

// =====================================================================
extern "C" void kernel_launch(void* const* d_in, const int* in_sizes, int n_in,
                              void* d_out, int out_size, void* d_ws, size_t ws_size,
                              hipStream_t stream) {
  (void)in_sizes; (void)n_in; (void)out_size; (void)ws_size;
  const float* q_in  = (const float*)d_in[0];
  const float* kv_in = (const float*)d_in[1];
  const float* mask  = (const float*)d_in[2];
  const float* Wq    = (const float*)d_in[3];
  const float* bq    = (const float*)d_in[4];
  const float* Wk    = (const float*)d_in[5];
  const float* Wv    = (const float*)d_in[6];
  const float* bv    = (const float*)d_in[7];
  const float* Wo    = (const float*)d_in[8];
  const float* bo    = (const float*)d_in[9];
  float* out = (float*)d_out;

  const size_t NM = (size_t)B * SQ * H;   // 4 M activations
  const size_t NW = (size_t)H * H;        // 1 M per weight

  _Float16* Xqh  = (_Float16*)d_ws;       // f16 staging
  _Float16* Xkvh = Xqh  + NM;
  _Float16* Wqh  = Xkvh + NM;
  _Float16* Wkh  = Wqh  + NW;
  _Float16* Wvh  = Wkh  + NW;
  _Float16* Woh  = Wvh  + NW;
  _Float16* Qh   = Woh  + NW;
  _Float16* Kh   = Qh   + NM;
  _Float16* VTh  = Kh   + NM;
  _Float16* ctx  = Xqh;                   // alias: Xq f16 dead after kernel 1

  dim3 blk(256);
  cvt_kernel<<<dim3(NM / 2048), blk, 0, stream>>>(q_in,  Xqh,  (int)NM);
  cvt_kernel<<<dim3(NM / 2048), blk, 0, stream>>>(kv_in, Xkvh, (int)NM);
  cvt_kernel<<<dim3(NW / 2048), blk, 0, stream>>>(Wq, Wqh, (int)NW);
  cvt_kernel<<<dim3(NW / 2048), blk, 0, stream>>>(Wk, Wkh, (int)NW);
  cvt_kernel<<<dim3(NW / 2048), blk, 0, stream>>>(Wv, Wvh, (int)NW);
  cvt_kernel<<<dim3(NW / 2048), blk, 0, stream>>>(Wo, Woh, (int)NW);

  qkv_proj_kernel<<<dim3((B * SQ) / 64, H / 256, 3), blk, 0, stream>>>(
      Xqh, Xkvh, Wqh, Wkh, Wvh, bq, bv, Qh, Kh, VTh);
  attn_kernel<<<dim3(SQ / 32, B * 2), blk, 0, stream>>>(Qh, Kh, VTh, mask, ctx);
  out_proj_kernel<<<dim3((B * SQ) / 64, H / 256), blk, 0, stream>>>(ctx, Woh, bo, out);
}
